// LSTMAutoencoderModel_85925115723839
// MI455X (gfx1250) — matrix operand order repaired and checked by reading, not compiled, not run on hardware
//
#include <hip/hip_runtime.h>
#include <hip/hip_bf16.h>

// LSTM autoencoder for MI455X (gfx1250, wave32, WMMA).
// Strategy: per-timestep fused (GEMM + gates) kernels using
// v_wmma_f32_16x16x32_bf16; weights converted once per call to bf16.
// Weight layout [4H][H] row-major is K-contiguous per output column,
// matching the WMMA B-operand fragment exactly -> pure b128 loads,
// no transposes, no LDS staging needed. z never leaves registers.

#define BB 128
#define SS 512
#define FF 64
#define HH 512
#define H4 2048

typedef __attribute__((ext_vector_type(16))) __bf16 v16bf;
typedef __attribute__((ext_vector_type(8)))  float  v8f;
typedef __attribute__((ext_vector_type(4)))  unsigned u32x4;

__device__ __forceinline__ v16bf frag_cast(u32x4 lo, u32x4 hi) {
    union { struct { u32x4 a, b; } s; v16bf v; } u;
    u.s.a = lo; u.s.b = hi;
    return u.v;
}

// A fragment (16x32 bf16, row-major source, stride ld elements).
// ISA layout: lanes 0-15 hold M=lane, K {k0..k0+7, k0+16..k0+23};
//             lanes 16-31 hold M=lane-16, K {k0+8..15, k0+24..31}.
__device__ __forceinline__ v16bf load_frag_A(const __bf16* __restrict__ a,
                                             int ld, int m0, int k0, int lane) {
    const int half = lane >> 4;
    const int row  = m0 + (lane & 15);
    const __bf16* p = a + (size_t)row * ld + k0 + half * 8;
    u32x4 lo = *reinterpret_cast<const u32x4*>(p);
    u32x4 hi = *reinterpret_cast<const u32x4*>(p + 16);
    return frag_cast(lo, hi);
}

// B fragment (32x16 bf16). Weights stored [N][K] row-major (stride ld):
// lane L owns column n0+(L&15); lanes 0-15 need K {k0..k0+7, k0+16..k0+23},
// lanes 16-31 need K {k0+8..k0+15, k0+24..k0+31} -> two b128 loads.
__device__ __forceinline__ v16bf load_frag_B(const __bf16* __restrict__ w,
                                             int ld, int n0, int k0, int lane) {
    const int half = lane >> 4;
    const int row  = n0 + (lane & 15);
    const __bf16* p = w + (size_t)row * ld + k0 + half * 8;
    u32x4 lo = *reinterpret_cast<const u32x4*>(p);
    u32x4 hi = *reinterpret_cast<const u32x4*>(p + 16);
    return frag_cast(lo, hi);
}

__device__ __forceinline__ float sigmoidf_(float x) {
    return 1.0f / (1.0f + __expf(-x));
}

// One LSTM cell step: z = X @ Wx^T + Hin @ Wh^T + bih + bhh; gate math;
// update C in place; write Hout (bf16) for the next GEMM.
// Grid: (HH/64, BB/32), 256 threads (8 waves); wave -> one 16x16 h-tile,
// 4 gate accumulators (i,f,g,o).
__global__ void __launch_bounds__(256)
lstm_cell_kernel(const __bf16* __restrict__ X, int ldx, int Kx,
                 const __bf16* __restrict__ Wx,   // [4H][Kx] bf16
                 const __bf16* __restrict__ Wh,   // [4H][HH] bf16
                 const float*  __restrict__ bih,
                 const float*  __restrict__ bhh,
                 const __bf16* __restrict__ Hin,  // [BB][HH] bf16
                 float*        __restrict__ Cst,  // [BB][HH] f32 (in/out)
                 __bf16*       __restrict__ Hout) // [BB][HH] bf16 (out)
{
    const int lane = threadIdx.x & 31;
    const int wv   = threadIdx.x >> 5;
    const int m0   = (blockIdx.y * 2 + (wv >> 2)) * 16; // batch tile
    const int n0   = (blockIdx.x * 4 + (wv & 3)) * 16;  // H-column tile
    const int col  = lane & 15;

    v8f acc[4];
#pragma unroll
    for (int g = 0; g < 4; ++g) {
        const int cz = g * HH + n0 + col;
        const float bz = bih[cz] + bhh[cz];
#pragma unroll
        for (int r = 0; r < 8; ++r) acc[g][r] = bz;
    }

    // Input contribution: K = Kx (64 or 512)
    for (int k0 = 0; k0 < Kx; k0 += 32) {
        v16bf a = load_frag_A(X, ldx, m0, k0, lane);
#pragma unroll
        for (int g = 0; g < 4; ++g) {
            v16bf b = load_frag_B(Wx, Kx, g * HH + n0, k0, lane);
            acc[g] = __builtin_amdgcn_wmma_f32_16x16x32_bf16(
                false, a, false, b, (short)0, acc[g], false, false);
        }
    }
    // Recurrent contribution: K = HH
    for (int k0 = 0; k0 < HH; k0 += 32) {
        v16bf a = load_frag_A(Hin, HH, m0, k0, lane);
#pragma unroll
        for (int g = 0; g < 4; ++g) {
            v16bf b = load_frag_B(Wh, HH, g * HH + n0, k0, lane);
            acc[g] = __builtin_amdgcn_wmma_f32_16x16x32_bf16(
                false, a, false, b, (short)0, acc[g], false, false);
        }
    }

    // Fused gate nonlinearities + state update.
    const int rbase = (lane >> 4) * 8; // M = r + 8*half per ISA C/D layout
#pragma unroll
    for (int r = 0; r < 8; ++r) {
        const int row = m0 + rbase + r;
        const int idx = row * HH + n0 + col;
        const float zi = acc[0][r], zf = acc[1][r];
        const float zg = acc[2][r], zo = acc[3][r];
        const float cn = sigmoidf_(zf) * Cst[idx] + sigmoidf_(zi) * tanhf(zg);
        const float hn = sigmoidf_(zo) * tanhf(cn);
        Cst[idx]  = cn;
        Hout[idx] = (__bf16)hn;
    }
}

// pred = h1 @ fcW^T + fc_b.  Writes f32 output slice (stride ldo) and the
// bf16 feedback buffer for the next decoder step.
// Grid: (BB/32), 256 threads; wave -> one 16x16 tile (N=64 = 4 waves wide).
__global__ void __launch_bounds__(256)
fc_kernel(const __bf16* __restrict__ Hin,   // [BB][HH]
          const __bf16* __restrict__ Wf,    // [FF][HH]
          const float*  __restrict__ bf,    // [FF]
          float*        __restrict__ Out,   // d_out + t*FF
          int ldo,                          // SS*FF
          __bf16*       __restrict__ Pred)  // [BB][FF]
{
    const int lane = threadIdx.x & 31;
    const int wv   = threadIdx.x >> 5;
    const int m0   = (blockIdx.x * 2 + (wv >> 2)) * 16;
    const int n0   = (wv & 3) * 16;
    const int col  = lane & 15;

    v8f acc;
    const float bz = bf[n0 + col];
#pragma unroll
    for (int r = 0; r < 8; ++r) acc[r] = bz;

    for (int k0 = 0; k0 < HH; k0 += 32) {
        v16bf a = load_frag_A(Hin, HH, m0, k0, lane);
        v16bf b = load_frag_B(Wf, HH, n0, k0, lane);
        acc = __builtin_amdgcn_wmma_f32_16x16x32_bf16(
            false, a, false, b, (short)0, acc, false, false);
    }

    const int rbase = (lane >> 4) * 8;
#pragma unroll
    for (int r = 0; r < 8; ++r) {
        const int row = m0 + rbase + r;
        const float v = acc[r];
        Out[(size_t)row * ldo + n0 + col] = v;
        Pred[row * FF + n0 + col] = (__bf16)v;
    }
}

__global__ void f32_to_bf16_kernel(const float* __restrict__ src,
                                   __bf16* __restrict__ dst, int n) {
    int i = blockIdx.x * blockDim.x + threadIdx.x;
    if (i < n) dst[i] = (__bf16)src[i];
}

__global__ void zero_u32_kernel(unsigned* __restrict__ p, int n) {
    int i = blockIdx.x * blockDim.x + threadIdx.x;
    if (i < n) p[i] = 0u;
}

extern "C" void kernel_launch(void* const* d_in, const int* in_sizes, int n_in,
                              void* d_out, int out_size, void* d_ws, size_t ws_size,
                              hipStream_t stream) {
    const float* x        = (const float*)d_in[0];
    const float* eWih0    = (const float*)d_in[1];
    const float* eWhh0    = (const float*)d_in[2];
    const float* ebih0    = (const float*)d_in[3];
    const float* ebhh0    = (const float*)d_in[4];
    const float* eWih1    = (const float*)d_in[5];
    const float* eWhh1    = (const float*)d_in[6];
    const float* ebih1    = (const float*)d_in[7];
    const float* ebhh1    = (const float*)d_in[8];
    const float* dWih0    = (const float*)d_in[9];
    const float* dWhh0    = (const float*)d_in[10];
    const float* dbih0    = (const float*)d_in[11];
    const float* dbhh0    = (const float*)d_in[12];
    const float* dWih1    = (const float*)d_in[13];
    const float* dWhh1    = (const float*)d_in[14];
    const float* dbih1    = (const float*)d_in[15];
    const float* dbhh1    = (const float*)d_in[16];
    const float* fcW      = (const float*)d_in[17];
    const float* fcb      = (const float*)d_in[18];
    float* out = (float*)d_out;

    // ---- workspace carve (all 256B aligned) ----
    char* ws = (char*)d_ws;
    size_t off = 0;
    auto carve = [&](size_t bytes) -> char* {
        char* p = ws + off;
        off += (bytes + 255) & ~(size_t)255;
        return p;
    };
    __bf16* x_bf   = (__bf16*)carve((size_t)BB * SS * FF * 2);
    __bf16* weWih0 = (__bf16*)carve((size_t)H4 * FF * 2);
    __bf16* weWhh0 = (__bf16*)carve((size_t)H4 * HH * 2);
    __bf16* weWih1 = (__bf16*)carve((size_t)H4 * HH * 2);
    __bf16* weWhh1 = (__bf16*)carve((size_t)H4 * HH * 2);
    __bf16* wdWih0 = (__bf16*)carve((size_t)H4 * FF * 2);
    __bf16* wdWhh0 = (__bf16*)carve((size_t)H4 * HH * 2);
    __bf16* wdWih1 = (__bf16*)carve((size_t)H4 * HH * 2);
    __bf16* wdWhh1 = (__bf16*)carve((size_t)H4 * HH * 2);
    __bf16* wfc    = (__bf16*)carve((size_t)FF * HH * 2);
    __bf16* h0buf[2] = { (__bf16*)carve((size_t)BB * HH * 2),
                         (__bf16*)carve((size_t)BB * HH * 2) };
    __bf16* h1buf[2] = { (__bf16*)carve((size_t)BB * HH * 2),
                         (__bf16*)carve((size_t)BB * HH * 2) };
    float*  c0     = (float*)carve((size_t)BB * HH * 4);
    float*  c1     = (float*)carve((size_t)BB * HH * 4);
    __bf16* pred   = (__bf16*)carve((size_t)BB * FF * 2);
    (void)ws_size; (void)in_sizes; (void)n_in; (void)out_size;

    auto conv = [&](const float* s, __bf16* d, int n) {
        f32_to_bf16_kernel<<<(n + 255) / 256, 256, 0, stream>>>(s, d, n);
    };
    // ---- prologue: weight/input conversion, state init ----
    conv(x,     x_bf,   BB * SS * FF);
    conv(eWih0, weWih0, H4 * FF);
    conv(eWhh0, weWhh0, H4 * HH);
    conv(eWih1, weWih1, H4 * HH);
    conv(eWhh1, weWhh1, H4 * HH);
    conv(dWih0, wdWih0, H4 * FF);
    conv(dWhh0, wdWhh0, H4 * HH);
    conv(dWih1, wdWih1, H4 * HH);
    conv(dWhh1, wdWhh1, H4 * HH);
    conv(fcW,   wfc,    FF * HH);

    const int hWords = BB * HH * 2 / 4; // bf16 buffer in u32 words
    const int cWords = BB * HH;
    zero_u32_kernel<<<(hWords + 255) / 256, 256, 0, stream>>>((unsigned*)h0buf[0], hWords);
    zero_u32_kernel<<<(hWords + 255) / 256, 256, 0, stream>>>((unsigned*)h1buf[0], hWords);
    zero_u32_kernel<<<(cWords + 255) / 256, 256, 0, stream>>>((unsigned*)c0, cWords);
    zero_u32_kernel<<<(cWords + 255) / 256, 256, 0, stream>>>((unsigned*)c1, cWords);

    const dim3 cellGrid(HH / 64, BB / 32); // 8 x 4
    int par = 0;

    // ---- encoder: layers interleaved step-by-step (same math as scan-then-scan)
    for (int t = 0; t < SS; ++t) {
        lstm_cell_kernel<<<cellGrid, 256, 0, stream>>>(
            x_bf + (size_t)t * FF, SS * FF, FF,
            weWih0, weWhh0, ebih0, ebhh0,
            h0buf[par], c0, h0buf[par ^ 1]);
        lstm_cell_kernel<<<cellGrid, 256, 0, stream>>>(
            h0buf[par ^ 1], HH, HH,
            weWih1, weWhh1, ebih1, ebhh1,
            h1buf[par], c1, h1buf[par ^ 1]);
        par ^= 1;
    }

    // ---- decoder: states continue from encoder finals in-place ----
    for (int t = 0; t < SS; ++t) {
        const __bf16* pin = (t == 0) ? (x_bf + (size_t)(SS - 1) * FF) : pred;
        const int ldin    = (t == 0) ? SS * FF : FF;
        lstm_cell_kernel<<<cellGrid, 256, 0, stream>>>(
            pin, ldin, FF,
            wdWih0, wdWhh0, dbih0, dbhh0,
            h0buf[par], c0, h0buf[par ^ 1]);
        lstm_cell_kernel<<<cellGrid, 256, 0, stream>>>(
            h0buf[par ^ 1], HH, HH,
            wdWih1, wdWhh1, dbih1, dbhh1,
            h1buf[par], c1, h1buf[par ^ 1]);
        fc_kernel<<<BB / 32, 256, 0, stream>>>(
            h1buf[par ^ 1], wfc, fcb,
            out + (size_t)t * FF, SS * FF, pred);
        par ^= 1;
    }
}
